// FG_4552665333921
// MI455X (gfx1250) — matrix operand order repaired and checked by reading, not compiled, not run on hardware
//
#include <hip/hip_runtime.h>
#include <hip/hip_bf16.h>

typedef _Float16 f16;
typedef __attribute__((ext_vector_type(16))) _Float16 v16h;
typedef __attribute__((ext_vector_type(8)))  _Float16 v8h;
typedef __attribute__((ext_vector_type(8)))  float    v8f;

typedef __attribute__((ext_vector_type(4))) unsigned int u32x4;
typedef __attribute__((ext_vector_type(8))) int          i32x8;
typedef __attribute__((ext_vector_type(4))) int          i32x4;

#define BB 32
#define HH 128
#define WW 128
#define NKP 49          // 7*7 glimpse positions
#define BN_EPS 1e-5f

#if __has_builtin(__builtin_amdgcn_tensor_load_to_lds) && __has_builtin(__builtin_amdgcn_s_wait_tensorcnt)
#define HAS_TDM 1
#else
#define HAS_TDM 0
#endif
#if __has_builtin(__builtin_amdgcn_global_load_async_to_lds_b128)
#define HAS_ASYNC 1
#else
#define HAS_ASYNC 0
#endif

__device__ __forceinline__ float eluf(float x)      { return x > 0.f ? x : expf(x) - 1.f; }
__device__ __forceinline__ float softplusf(float x) { return fmaxf(x, 0.f) + log1pf(expf(-fabsf(x))); }
__device__ __forceinline__ float sigmoidf(float x)  { return 1.f / (1.f + expf(-x)); }

// ---------------------------------------------------------------------------
// Layer 1: direct conv 3->64, stride 2, VALID. Channels-last f16 out + BN stats.
// ---------------------------------------------------------------------------
__global__ void conv1_kernel(const float* __restrict__ img, const float* __restrict__ w,
                             const float* __restrict__ bias, f16* __restrict__ out,
                             float* __restrict__ stats)
{
    __shared__ float ssum[64], ssq[64];
    int tid = threadIdx.x;
    if (tid < 64) { ssum[tid] = 0.f; ssq[tid] = 0.f; }
    __syncthreads();

    int idx = blockIdx.x * blockDim.x + tid;      // 32*63*63*64, exact multiple of 256
    int c = idx & 63;
    int p = idx >> 6;
    int b = p / (63 * 63);
    int rem = p % (63 * 63);
    int oy = rem / 63, ox = rem % 63;

    float v = bias[c];
    #pragma unroll
    for (int ci = 0; ci < 3; ++ci)
        #pragma unroll
        for (int ky = 0; ky < 3; ++ky)
            #pragma unroll
            for (int kx = 0; kx < 3; ++kx)
                v += img[((b * 3 + ci) * HH + oy * 2 + ky) * WW + ox * 2 + kx] *
                     w[((c * 3 + ci) * 3 + ky) * 3 + kx];
    out[idx] = (f16)v;
    atomicAdd(&ssum[c], v);
    atomicAdd(&ssq[c], v * v);
    __syncthreads();
    if (tid < 64) {
        atomicAdd(&stats[tid], ssum[tid]);
        atomicAdd(&stats[64 + tid], ssq[tid]);
    }
}

__global__ void bn_finalize_kernel(const float* __restrict__ stats, float* __restrict__ bn, float invN)
{
    int c = threadIdx.x;
    float mean = stats[c] * invN;
    float var  = stats[64 + c] * invN - mean * mean;
    bn[c]      = mean;
    bn[64 + c] = rsqrtf(var + BN_EPS);
}

__global__ void bn_elu_kernel(f16* __restrict__ act, const float* __restrict__ bn,
                              const float* __restrict__ g, const float* __restrict__ be, int total)
{
    int idx = blockIdx.x * blockDim.x + threadIdx.x;
    if (idx >= total) return;
    int c = idx & 63;
    float x = (float)act[idx];
    float y = g[c] * (x - bn[c]) * bn[64 + c] + be[c];
    act[idx] = (f16)eluf(y);
}

// ---------------------------------------------------------------------------
// Pack OIHW f32 weights (64,64,3,3) into [18][64][32] f16 slabs:
// k = ky*192 + kx*64 + c ; slab s covers k in [32s, 32s+32), stored [s][n][kk].
// ---------------------------------------------------------------------------
__global__ void pack_weights_kernel(const float* __restrict__ w, f16* __restrict__ dst)
{
    int idx = blockIdx.x * blockDim.x + threadIdx.x;   // 18*64*32 = 36864
    int kk = idx & 31;
    int n  = (idx >> 5) & 63;
    int s  = idx >> 11;
    int c  = ((s & 1) << 5) + kk;
    int kx = (s >> 1) % 3;
    int ky = s / 6;
    dst[idx] = (f16)w[((n * 64 + c) * 3 + ky) * 3 + kx];
}

// ---------------------------------------------------------------------------
// WMMA implicit-GEMM conv: 64->64, 3x3 stride 2 VALID. K = 576 = 3 phases x 6
// slabs x 32. Weights staged into LDS per phase via TDM (fallback: async-LDS,
// plain). A-fragments loaded directly from global in VGPR-layout order.
// One wave = 16 pixels x 64 out-channels; 2 waves/block; tiles/2 exact.
// ---------------------------------------------------------------------------
__global__ void __launch_bounds__(64) conv_wmma_kernel(
    const f16* __restrict__ in, const f16* __restrict__ wpack,
    const float* __restrict__ bias, f16* __restrict__ out,
    float* __restrict__ stats, int Hin, int Win, int Hout, int Wout)
{
    __shared__ __align__(16) f16 Wlds[6][64][32];   // 24 KB: one ky-row of K-slabs

    const int tid  = threadIdx.x;
    const int lane = tid & 31;
    const int wv   = tid >> 5;
    const int T    = blockIdx.x * 2 + wv;
    const int HWo  = Hout * Wout;
    const int mrow = lane & 15;
    const int half = lane >> 4;

    // This lane's A pixel row (M = mrow), base of its channels-last row.
    int p = T * 16 + mrow;
    int b = p / HWo, rem = p % HWo;
    int oy = rem / Wout, ox = rem % Wout;
    const f16* abase = in + ((b * Hin + oy * 2) * Win + ox * 2) * 64;

    v8f acc[4] = {v8f{}, v8f{}, v8f{}, v8f{}};

    for (int ph = 0; ph < 3; ++ph) {               // ph == ky
        const f16* wsrc = wpack + ph * (6 * 64 * 32);

        // ---- Stage 24 KB of packed weights into LDS ----
#if HAS_TDM
        if (wv == 0) {
            unsigned lds_base = (unsigned)(size_t)&Wlds[0][0][0];
            unsigned long long ga = (unsigned long long)(size_t)wsrc;
            u32x4 g0;
            g0[0] = 1u;                                  // count=1 user descriptor
            g0[1] = lds_base;                            // lds_addr
            g0[2] = (unsigned)ga;                        // global_addr[31:0]
            g0[3] = (unsigned)(ga >> 32) | (2u << 30);   // global_addr[56:32] | type=2
            i32x8 g1;
            g1[0] = (3 << 16);                           // workgroup_mask=0, data_size=8B
            g1[1] = (int)(3072u << 16);                  // tensor_dim0[15:0] (3072 x 8B)
            g1[2] = (int)(1u << 16);                     // tensor_dim0 hi=0 | tensor_dim1=1
            g1[3] = (int)(3072u << 16);                  // tensor_dim1 hi=0 | tile_dim0=3072
            g1[4] = 1;                                   // tile_dim1=1, tile_dim2=0
            g1[5] = 3072;                                // tensor_dim0_stride low32
            g1[6] = 0;
            g1[7] = 0;
            i32x4 gz = {0, 0, 0, 0};
            i32x8 gz8 = {0, 0, 0, 0, 0, 0, 0, 0};
            __builtin_amdgcn_tensor_load_to_lds(g0, g1, gz, gz, gz8, 0);
            __builtin_amdgcn_s_wait_tensorcnt(0);
        }
        __syncthreads();
#elif HAS_ASYNC
        for (int q = tid; q < 1536; q += 64) {           // 1536 x 16B = 24 KB
            __builtin_amdgcn_global_load_async_to_lds_b128(
                (__attribute__((address_space(1))) const void*)(size_t)(wsrc + q * 8),
                (__attribute__((address_space(3))) void*)(unsigned)(size_t)(&Wlds[0][0][0] + q * 8),
                0, 0);
        }
        asm volatile("s_wait_asynccnt 0" ::: "memory");
        __syncthreads();
#else
        {
            const v8h* srcv = (const v8h*)wsrc;
            v8h* dstv = (v8h*)&Wlds[0][0][0];
            for (int q = tid; q < 1536; q += 64) dstv[q] = srcv[q];
        }
        __syncthreads();
#endif
        if (ph + 1 < 3) __builtin_prefetch(wpack + (ph + 1) * (6 * 64 * 32), 0, 0);

        // ---- 6 K-slabs: kx = ss>>1, channel half = ss&1 ----
        #pragma unroll
        for (int ss = 0; ss < 6; ++ss) {
            const int kx = ss >> 1, chalf = (ss & 1) << 5;
            const f16* arow = abase + (ph * Win + kx) * 64 + chalf;

            // A fragment: lanes 0-15 hold K 0..15, lanes 16-31 K 16..31 (matches B).
            v16h afrag = *(const v16h*)(arow + 16 * half);

            // B fragment: lanes 0-15 hold K 0..15, lanes 16-31 K 16..31 (N = lane&15).
            #pragma unroll
            for (int nt = 0; nt < 4; ++nt) {
                v16h bfrag = *(const v16h*)(&Wlds[ss][nt * 16 + mrow][half * 16]);
                acc[nt] = __builtin_amdgcn_wmma_f32_16x16x32_f16(
                    false, afrag, false, bfrag, (short)0, acc[nt], false, false);
            }
        }
        __syncthreads();   // before next phase overwrites Wlds
    }

    // Epilogue: bias, store f16 channels-last, accumulate BN stats.
    // C/D layout: N = lane&15 ; VGPR r -> M = r + 8*(lane>>4).
    float bs[4] = {0.f, 0.f, 0.f, 0.f}, bq[4] = {0.f, 0.f, 0.f, 0.f};
    #pragma unroll
    for (int r = 0; r < 8; ++r) {
        int m = half * 8 + r;
        int obase = (T * 16 + m) * 64;
        #pragma unroll
        for (int nt = 0; nt < 4; ++nt) {
            int n = nt * 16 + mrow;
            float v = acc[nt][r] + bias[n];
            out[obase + n] = (f16)v;
            bs[nt] += v;
            bq[nt] += v * v;
        }
    }
    #pragma unroll
    for (int nt = 0; nt < 4; ++nt) {
        int n = nt * 16 + mrow;
        atomicAdd(&stats[n], bs[nt]);
        atomicAdd(&stats[64 + n], bq[nt]);
    }
}

// ---------------------------------------------------------------------------
// Head: 1x1 conv 64->64 + ELU, 1x1 conv 64->11, glimpse params.
// ---------------------------------------------------------------------------
__global__ void head_kernel(const f16* __restrict__ act4,
                            const float* __restrict__ wp, const float* __restrict__ bias_p,
                            const float* __restrict__ wz, const float* __restrict__ bz,
                            const float* __restrict__ eps_scale, const float* __restrict__ eps_shift,
                            float* __restrict__ params)
{
    __shared__ float av[64], fv[64], zb[11];
    int tid = threadIdx.x;
    int bpx = blockIdx.x;                 // b*49 + pix
    int b = bpx / NKP, pix = bpx % NKP;

    av[tid] = (float)act4[bpx * 64 + tid];
    __syncthreads();

    float s = bias_p[tid];
    #pragma unroll 8
    for (int c = 0; c < 64; ++c) s += av[c] * wp[tid * 64 + c];
    fv[tid] = eluf(s);
    __syncthreads();

    if (tid < 11) {
        float z = bz[tid];
        #pragma unroll 8
        for (int c = 0; c < 64; ++c) z += fv[c] * wz[tid * 64 + c];
        zb[tid] = z;
    }
    __syncthreads();

    if (tid < 2) {
        int comp = tid;
        int i = pix / 7, j = pix % 7;
        float es = eps_scale[((b * 7 + i) * 7 + j) * 2 + comp];
        float eh = eps_shift[((b * 7 + i) * 7 + j) * 2 + comp];
        float zsc = zb[3 + comp] + softplusf(zb[5 + comp]) * es;
        float zsh = zb[7 + comp] + softplusf(zb[9 + comp]) * eh;
        float scale  = sigmoidf(zsc) * 0.25f;          // GH/H = GW/W = 32/128
        float shift  = sigmoidf(zsh) * (1.f / 7.f);
        float ijc    = (comp == 0 ? (float)i : (float)j) * (1.f / 7.f);
        float center = (ijc + shift) * 2.f - 1.f;
        params[comp * (BB * NKP) + bpx]       = scale;   // th / tw
        params[(2 + comp) * (BB * NKP) + bpx] = center;  // tx / ty
    }
}

// ---------------------------------------------------------------------------
// Bilinear glimpse sampler. Output (B*49, 3, 32, 32) f32.
// ---------------------------------------------------------------------------
__global__ void sample_kernel(const float* __restrict__ img, const float* __restrict__ params,
                              float* __restrict__ out)
{
    int idx = blockIdx.x * blockDim.x + threadIdx.x;
    const int total = BB * NKP * 32 * 32;
    if (idx >= total) return;
    int gx = idx & 31;
    int gy = (idx >> 5) & 31;
    int bk = idx >> 10;                 // b*49 + k
    int b  = bk / NKP;

    const int NP = BB * NKP;
    float th = params[bk], tw = params[NP + bk];
    float tx = params[2 * NP + bk], ty = params[3 * NP + bk];

    float xb = (2.f * gx + 1.f) * (1.f / 32.f) - 1.f;
    float yb = (2.f * gy + 1.f) * (1.f / 32.f) - 1.f;
    float ixn = tw * xb + ty;
    float iyn = th * yb + tx;
    float px = ((ixn + 1.f) * WW - 1.f) * 0.5f;
    float py = ((iyn + 1.f) * HH - 1.f) * 0.5f;

    float x0 = floorf(px), y0 = floorf(py);
    float wx1 = px - x0, wy1 = py - y0;
    int x0i = (int)x0, y0i = (int)y0, x1i = x0i + 1, y1i = y0i + 1;
    float w00 = (1.f - wy1) * (1.f - wx1), w01 = (1.f - wy1) * wx1;
    float w10 = wy1 * (1.f - wx1),         w11 = wy1 * wx1;
    int x0c = min(max(x0i, 0), WW - 1), x1c = min(max(x1i, 0), WW - 1);
    int y0c = min(max(y0i, 0), HH - 1), y1c = min(max(y1i, 0), HH - 1);
    float m00 = (x0i >= 0 && x0i < WW && y0i >= 0 && y0i < HH) ? 1.f : 0.f;
    float m01 = (x1i >= 0 && x1i < WW && y0i >= 0 && y0i < HH) ? 1.f : 0.f;
    float m10 = (x0i >= 0 && x0i < WW && y1i >= 0 && y1i < HH) ? 1.f : 0.f;
    float m11 = (x1i >= 0 && x1i < WW && y1i >= 0 && y1i < HH) ? 1.f : 0.f;

    #pragma unroll
    for (int c = 0; c < 3; ++c) {
        const float* im = img + (b * 3 + c) * HH * WW;
        float v = im[y0c * WW + x0c] * (m00 * w00) + im[y0c * WW + x1c] * (m01 * w01)
                + im[y1c * WW + x0c] * (m10 * w10) + im[y1c * WW + x1c] * (m11 * w11);
        out[(bk * 3 + c) * 1024 + (gy << 5) + gx] = v;
    }
}

// ---------------------------------------------------------------------------
extern "C" void kernel_launch(void* const* d_in, const int* in_sizes, int n_in,
                              void* d_out, int out_size, void* d_ws, size_t ws_size,
                              hipStream_t stream)
{
    (void)in_sizes; (void)n_in; (void)out_size; (void)ws_size;
    const float* img       = (const float*)d_in[0];
    const float* eps_scale = (const float*)d_in[1];
    const float* eps_shift = (const float*)d_in[2];
    const float *w[4], *bia[4], *g[4], *be[4];
    for (int i = 0; i < 4; ++i) {
        w[i]   = (const float*)d_in[3 + i * 4];
        bia[i] = (const float*)d_in[4 + i * 4];
        g[i]   = (const float*)d_in[5 + i * 4];
        be[i]  = (const float*)d_in[6 + i * 4];
    }
    const float* wp  = (const float*)d_in[19];
    const float* bpv = (const float*)d_in[20];
    const float* wz  = (const float*)d_in[21];
    const float* bz  = (const float*)d_in[22];
    float* outp = (float*)d_out;

    // Workspace carve-up (256B aligned).
    char* ws = (char*)d_ws;
    size_t off = 0;
    auto carve = [&](size_t bytes) -> char* {
        char* p = ws + off;
        off = (off + bytes + 255) & ~(size_t)255;
        return p;
    };
    float* stats  = (float*)carve(4 * 128 * sizeof(float));   // [layer][sum|sumsq][64]
    float* bn     = (float*)carve(4 * 128 * sizeof(float));   // [layer][mean|rstd][64]
    f16*   wpack  = (f16*)carve((size_t)3 * 18 * 64 * 32 * sizeof(f16));
    f16*   act1   = (f16*)carve((size_t)BB * 63 * 63 * 64 * sizeof(f16));
    f16*   act2   = (f16*)carve((size_t)BB * 31 * 31 * 64 * sizeof(f16));
    f16*   act3   = (f16*)carve((size_t)BB * 15 * 15 * 64 * sizeof(f16));
    f16*   act4   = (f16*)carve((size_t)BB * 7 * 7 * 64 * sizeof(f16));
    float* params = (float*)carve((size_t)4 * BB * NKP * sizeof(float));

    (void)hipMemsetAsync(stats, 0, 4 * 128 * sizeof(float), stream);

    // Pack weights for WMMA layers 2..4.
    for (int L = 0; L < 3; ++L)
        pack_weights_kernel<<<(18 * 64 * 32) / 256, 256, 0, stream>>>(w[1 + L], wpack + L * (18 * 64 * 32));

    // Layer 1 (direct) + BN + ELU.
    {
        int total = BB * 63 * 63 * 64;
        conv1_kernel<<<total / 256, 256, 0, stream>>>(img, w[0], bia[0], act1, stats);
        bn_finalize_kernel<<<1, 64, 0, stream>>>(stats, bn, 1.f / (float)(BB * 63 * 63));
        bn_elu_kernel<<<(total + 255) / 256, 256, 0, stream>>>(act1, bn, g[0], be[0], total);
    }

    // Layers 2..4 (WMMA implicit GEMM) + BN + ELU.
    struct Lyr { f16* in; f16* out; int Hin, Win, Hout, Wout; };
    Lyr Ls[3] = { { act1, act2, 63, 63, 31, 31 },
                  { act2, act3, 31, 31, 15, 15 },
                  { act3, act4, 15, 15, 7, 7 } };
    for (int L = 0; L < 3; ++L) {
        int tiles = BB * Ls[L].Hout * Ls[L].Wout / 16;   // even for all layers
        conv_wmma_kernel<<<tiles / 2, 64, 0, stream>>>(
            Ls[L].in, wpack + L * (18 * 64 * 32), bia[1 + L], Ls[L].out,
            stats + (1 + L) * 128, Ls[L].Hin, Ls[L].Win, Ls[L].Hout, Ls[L].Wout);
        int cnt = BB * Ls[L].Hout * Ls[L].Wout;
        bn_finalize_kernel<<<1, 64, 0, stream>>>(stats + (1 + L) * 128, bn + (1 + L) * 128, 1.f / (float)cnt);
        bn_elu_kernel<<<(cnt * 64 + 255) / 256, 256, 0, stream>>>(Ls[L].out, bn + (1 + L) * 128, g[1 + L], be[1 + L], cnt * 64);
    }

    // Head (1x1 convs + glimpse params) and bilinear sampler.
    head_kernel<<<BB * NKP, 64, 0, stream>>>(act4, wp, bpv, wz, bz, eps_scale, eps_shift, params);
    int stotal = BB * NKP * 32 * 32;
    sample_kernel<<<(stotal + 255) / 256, 256, 0, stream>>>(img, params, outp);
}
